// ADRNN_Linear_4063039062463
// MI455X (gfx1250) — matrix operand order repaired and checked by reading, not compiled, not run on hardware
//
#include <hip/hip_runtime.h>

// ADRNN (dual 2-layer LSTM scan, B=1024, T=512, R=47, ENC=32, T_SZ=2) for gfx1250.
// 64 single-wave (wave32) workgroups, one 16-row batch tile each, scanning all
// 512 steps. The scan is latency-bound (only 64 independent waves exist), so we
// spend VGPRs freely: ALL weight WMMA B-fragments (84 x v16h = 672 VGPRs) and
// bias lanes are preloaded into registers before the time loop; per-step WMMAs
// run against register-resident B operands. LDS holds only activation staging
// (A-fragment sources), gate dumps, c-state, and the tiny VALU t-path.

#define DEVI __device__ __forceinline__

typedef __attribute__((ext_vector_type(16))) _Float16 v16h;
typedef __attribute__((ext_vector_type(8)))  _Float16 v8h;
typedef __attribute__((ext_vector_type(8)))  float    v8f;

namespace {

constexpr int B_ALL = 1024;
constexpr int T_LEN = 512;
constexpr int RH    = 47;    // r hidden
constexpr int TS    = 2;     // t hidden
constexpr int GATES = 4 * RH;   // 188
constexpr int GP    = 192;      // gates padded to 12 WMMA N-tiles
constexpr int MB    = 16;       // batch rows per workgroup (one WMMA M-tile)

struct __align__(16) SMem {
  // --- weight staging (f16; only read during the one-time fragment preload) ---
  _Float16 e1w[32][64];    // e1_w [32x47] -> K padded to 64
  _Float16 e3w[32][64];    // e3_w [32x64]
  _Float16 e4w[32][64];    // e4_w [32x64]
  _Float16 wr0[GP][96];    // [wih0 (K 0..31) | whh0 (K 32..78)] rows padded 188->192
  _Float16 wr1[GP][96];    // [wih1 (K 0..46) | whh1 (K 48..94)]
  // --- per-step activation staging (f16, row-major, A-fragment source) ---
  _Float16 xr[MB][64];     // x_r tile, cols 47..63 = 0
  _Float16 catr[MB][64];   // [r_pre | t_in]
  _Float16 catt[MB][64];   // [t_in | r_in]
  _Float16 gin0[MB][96];   // [in_r (0..31) | hr0 (32..78)]
  _Float16 gin1[MB][128];  // [hr0 (0..46) | hr1 (48..94) | zeros]
  // --- f32 buffers ---
  float gates[MB][GP];     // gate GEMM dump (i:0..46, f:47..93, g:94..140, o:141..187)
  float itbuf[MB][32];     // in_t
  float cr0[MB][48];
  float cr1[MB][48];
  float e1b[32], e3b[32], e4b[32];
  float rb0[GP], rb1[GP];
  float e2w[32][2], e2b[32];
  float tw0[8][32], twh0[8][2], tb0[8];
  float twi1[8][2], twh1[8][2], tb1[8];
  float xt[32];            // x_t tile [16][2]
  float tg[MB][8];
  float ht0[MB][2], ct0[MB][2], ht1[MB][2], ct1[MB][2];
};

// ---- WMMA fragment helpers (CDNA5 16x16x32 f16, wave32 layouts per ISA 7.12.2) ----

// A (16xK slice of 32): lanes 0-15 hold M=0..15 with K = kbase + {0..7, 16..23},
// lanes 16-31 hold K = kbase + {8..15, 24..31}.
DEVI v16h load_A(const _Float16* base, int ld, int kbase) {
  const int lane = threadIdx.x & 31;
  const _Float16* p = base + (lane & 15) * ld + kbase + ((lane >> 4) << 3);
  v8h lo = *(const v8h*)(p);
  v8h hi = *(const v8h*)(p + 16);
  v16h a;
#pragma unroll
  for (int j = 0; j < 8; ++j) { a[j] = lo[j]; a[j + 8] = hi[j]; }
  return a;
}

// B (32x16 slice): lane holds column N = nbase + (lane&15); lanes 0-15 cover
// K = kbase + {0..7, 16..23}, lanes 16-31 cover K = kbase + {8..15, 24..31}.
DEVI v16h load_B(const _Float16* base, int ld, int nbase, int kbase) {
  const int lane = threadIdx.x & 31;
  const _Float16* p = base + (nbase + (lane & 15)) * ld + kbase + ((lane >> 4) << 3);
  v8h lo = *(const v8h*)(p);
  v8h hi = *(const v8h*)(p + 16);
  v16h b;
#pragma unroll
  for (int j = 0; j < 8; ++j) { b[j] = lo[j]; b[j + 8] = hi[j]; }
  return b;
}

// C/D accumulator layout: element (M = r + 8*(lane>>4), N = lane&15).
DEVI v8f splat(float v) {
  v8f c;
#pragma unroll
  for (int j = 0; j < 8; ++j) c[j] = v;
  return c;
}

DEVI v8f wmma(v16h a, v16h b, v8f c) {
  // (neg_a, A, neg_b, B, c_mod, C, reuse_a, reuse_b)
  return __builtin_amdgcn_wmma_f32_16x16x32_f16(false, a, false, b, (short)0, c,
                                                false, false);
}

DEVI void store_D_h(_Float16* base, int ld, int nbase, v8f d) {
  const int lane = threadIdx.x & 31;
  const int n = nbase + (lane & 15);
  const int mo = (lane >> 4) << 3;
#pragma unroll
  for (int r = 0; r < 8; ++r) base[(mo + r) * ld + n] = (_Float16)d[r];
}

DEVI void store_D_f(float* base, int ld, int nbase, v8f d) {
  const int lane = threadIdx.x & 31;
  const int n = nbase + (lane & 15);
  const int mo = (lane >> 4) << 3;
#pragma unroll
  for (int r = 0; r < 8; ++r) base[(mo + r) * ld + n] = d[r];
}

DEVI float sigf(float x) { return 1.0f / (1.0f + __expf(-x)); }

__global__ void __launch_bounds__(32, 1)
adrnn_scan_kernel(const float* __restrict__ x_r, const float* __restrict__ x_t,
                  const float* e1_w, const float* e1_b,
                  const float* e2_w, const float* e2_b,
                  const float* e3_w, const float* e3_b,
                  const float* e4_w, const float* e4_b,
                  const float* r_wih0, const float* r_whh0, const float* r_bih0, const float* r_bhh0,
                  const float* r_wih1, const float* r_whh1, const float* r_bih1, const float* r_bhh1,
                  const float* t_wih0, const float* t_whh0, const float* t_bih0, const float* t_bhh0,
                  const float* t_wih1, const float* t_whh1, const float* t_bih1, const float* t_bhh1,
                  float* __restrict__ outR, float* __restrict__ outT) {
  extern __shared__ char smraw[];
  SMem& sm = *reinterpret_cast<SMem*>(smraw);
  const int lane = threadIdx.x & 31;
  const int b0 = blockIdx.x * MB;

  // Zero all of LDS (also zero-initializes h/c state and pad columns).
  for (int i = lane; i < (int)(sizeof(SMem) / 4); i += 32)
    reinterpret_cast<int*>(smraw)[i] = 0;
  __syncthreads();

  // ---- One-time weight staging: f32 HBM -> f16 LDS ----
  for (int i = lane; i < 32 * 47; i += 32) {
    int n = i / 47, k = i - n * 47;
    sm.e1w[n][k] = (_Float16)e1_w[i];
  }
  for (int i = lane; i < 32 * 64; i += 32) {
    sm.e3w[i >> 6][i & 63] = (_Float16)e3_w[i];
    sm.e4w[i >> 6][i & 63] = (_Float16)e4_w[i];
  }
  for (int i = lane; i < GATES * 32; i += 32) {
    sm.wr0[i >> 5][i & 31] = (_Float16)r_wih0[i];
  }
  for (int i = lane; i < GATES * 47; i += 32) {
    int n = i / 47, k = i - n * 47;
    sm.wr0[n][32 + k] = (_Float16)r_whh0[i];
    sm.wr1[n][k]      = (_Float16)r_wih1[i];
    sm.wr1[n][48 + k] = (_Float16)r_whh1[i];
  }
  for (int i = lane; i < 32; i += 32) {
    sm.e1b[i] = e1_b[i]; sm.e3b[i] = e3_b[i]; sm.e4b[i] = e4_b[i];
    sm.e2b[i] = e2_b[i];
    sm.e2w[i][0] = e2_w[2 * i]; sm.e2w[i][1] = e2_w[2 * i + 1];
  }
  for (int i = lane; i < GATES; i += 32) {
    sm.rb0[i] = r_bih0[i] + r_bhh0[i];
    sm.rb1[i] = r_bih1[i] + r_bhh1[i];
  }
  for (int i = lane; i < 8 * 32; i += 32) sm.tw0[i >> 5][i & 31] = t_wih0[i];
  for (int i = lane; i < 16; i += 32) {
    sm.twh0[i >> 1][i & 1] = t_whh0[i];
    sm.twi1[i >> 1][i & 1] = t_wih1[i];
    sm.twh1[i >> 1][i & 1] = t_whh1[i];
  }
  for (int i = lane; i < 8; i += 32) {
    sm.tb0[i] = t_bih0[i] + t_bhh0[i];
    sm.tb1[i] = t_bih1[i] + t_bhh1[i];
  }
  __syncthreads();

  // ---- One-time fragment preload: ALL weight B-fragments + bias lanes -> VGPRs ----
  // (time-invariant; occupancy is 1 wave/SIMD anyway, so the 1024-VGPR file is free)
  v16h fe1[2][2], fe3[2][2], fe4[2][2];  // [ntile][kslice]
  v16h fw0[12][3], fw1[12][3];
  float be1[2], be3[2], be4[2], brb0[12], brb1[12];
#pragma unroll
  for (int nt = 0; nt < 2; ++nt) {
#pragma unroll
    for (int k = 0; k < 2; ++k) {
      fe1[nt][k] = load_B(&sm.e1w[0][0], 64, nt * 16, k * 32);
      fe3[nt][k] = load_B(&sm.e3w[0][0], 64, nt * 16, k * 32);
      fe4[nt][k] = load_B(&sm.e4w[0][0], 64, nt * 16, k * 32);
    }
    be1[nt] = sm.e1b[nt * 16 + (lane & 15)];
    be3[nt] = sm.e3b[nt * 16 + (lane & 15)];
    be4[nt] = sm.e4b[nt * 16 + (lane & 15)];
  }
#pragma unroll
  for (int nt = 0; nt < 12; ++nt) {
#pragma unroll
    for (int k = 0; k < 3; ++k) {
      fw0[nt][k] = load_B(&sm.wr0[0][0], 96, nt * 16, k * 32);
      fw1[nt][k] = load_B(&sm.wr1[0][0], 96, nt * 16, k * 32);
    }
    brb0[nt] = sm.rb0[nt * 16 + (lane & 15)];
    brb1[nt] = sm.rb1[nt * 16 + (lane & 15)];
  }
  __syncthreads();

  // ---- Sequential scan over time; batch tile stays on this wave ----
  for (int t = 0; t < T_LEN; ++t) {
    // Prefetch next step's x_r rows into cache (global_prefetch path).
    if (t + 1 < T_LEN) {
      for (int m = lane; m < MB; m += 32)
        __builtin_prefetch(&x_r[((long)(b0 + m) * T_LEN + (t + 1)) * RH], 0, 0);
    }
    // Stage x_t tile [16][2].
    {
      int m = lane >> 1, k = lane & 1;
      sm.xt[lane] = x_t[((long)(b0 + m) * T_LEN + t) * TS + k];
    }
    // Stage x_r tile [16][47] as f16 (cols 47..63 stay zero).
    for (int i = lane; i < MB * RH; i += 32) {
      int m = i / RH, k = i - m * RH;
      sm.xr[m][k] = (_Float16)x_r[((long)(b0 + m) * T_LEN + t) * RH + k];
    }
    __syncthreads();

    // t_in = x_t @ e2^T + e2_b  (K=2: VALU) -> catr[:,32:64] and catt[:,0:32]
    for (int i = lane; i < MB * 32; i += 32) {
      int m = i >> 5, n = i & 31;
      float v = sm.xt[2 * m] * sm.e2w[n][0] + sm.xt[2 * m + 1] * sm.e2w[n][1] + sm.e2b[n];
      sm.catr[m][32 + n] = (_Float16)v;
      sm.catt[m][n]      = (_Float16)v;
    }

    // r_pre = x_r @ e1^T + e1_b -> catr[:,0:32]
    {
      v16h a0 = load_A(&sm.xr[0][0], 64, 0);
      v16h a1 = load_A(&sm.xr[0][0], 64, 32);
#pragma unroll
      for (int nt = 0; nt < 2; ++nt) {
        v8f acc = splat(be1[nt]);
        acc = wmma(a0, fe1[nt][0], acc);
        acc = wmma(a1, fe1[nt][1], acc);
        store_D_h(&sm.catr[0][0], 64, nt * 16, acc);
      }
    }
    __syncthreads();

    // in_r = [r_pre | t_in] @ e3^T + e3_b -> gin0[:,0:32]
    {
      v16h a0 = load_A(&sm.catr[0][0], 64, 0);
      v16h a1 = load_A(&sm.catr[0][0], 64, 32);
#pragma unroll
      for (int nt = 0; nt < 2; ++nt) {
        v8f acc = splat(be3[nt]);
        acc = wmma(a0, fe3[nt][0], acc);
        acc = wmma(a1, fe3[nt][1], acc);
        store_D_h(&sm.gin0[0][0], 96, nt * 16, acc);
      }
    }
    __syncthreads();

    // r-cell 0 gates: [in_r | hr0] (16x96) @ wr0^T (96x192) + rb0
    {
      v16h a0 = load_A(&sm.gin0[0][0], 96, 0);
      v16h a1 = load_A(&sm.gin0[0][0], 96, 32);
      v16h a2 = load_A(&sm.gin0[0][0], 96, 64);
#pragma unroll
      for (int nt = 0; nt < 12; ++nt) {
        v8f acc = splat(brb0[nt]);
        acc = wmma(a0, fw0[nt][0], acc);
        acc = wmma(a1, fw0[nt][1], acc);
        acc = wmma(a2, fw0[nt][2], acc);
        store_D_f(&sm.gates[0][0], GP, nt * 16, acc);
      }
    }
    __syncthreads();

    // r-cell 0 elementwise update -> hr0 (gin0[:,32:79] for next step, gin1[:,0:47] now)
    for (int e = lane; e < MB * RH; e += 32) {
      int m = e / RH, j = e - m * RH;
      float cn = sigf(sm.gates[m][RH + j]) * sm.cr0[m][j]
               + sigf(sm.gates[m][j]) * tanhf(sm.gates[m][2 * RH + j]);
      float hn = sigf(sm.gates[m][3 * RH + j]) * tanhf(cn);
      sm.cr0[m][j] = cn;
      _Float16 hh = (_Float16)hn;
      sm.gin0[m][32 + j] = hh;
      sm.gin1[m][j]      = hh;
    }
    __syncthreads();

    // r-cell 1 gates: [hr0 | hr1] (16x96 of ld=128) @ wr1^T + rb1
    {
      v16h a0 = load_A(&sm.gin1[0][0], 128, 0);
      v16h a1 = load_A(&sm.gin1[0][0], 128, 32);
      v16h a2 = load_A(&sm.gin1[0][0], 128, 64);
#pragma unroll
      for (int nt = 0; nt < 12; ++nt) {
        v8f acc = splat(brb1[nt]);
        acc = wmma(a0, fw1[nt][0], acc);
        acc = wmma(a1, fw1[nt][1], acc);
        acc = wmma(a2, fw1[nt][2], acc);
        store_D_f(&sm.gates[0][0], GP, nt * 16, acc);
      }
    }
    __syncthreads();

    // r-cell 1 update -> hr1 (gin1[:,48:95]) + write r_out to HBM (f32)
    for (int e = lane; e < MB * RH; e += 32) {
      int m = e / RH, j = e - m * RH;
      float cn = sigf(sm.gates[m][RH + j]) * sm.cr1[m][j]
               + sigf(sm.gates[m][j]) * tanhf(sm.gates[m][2 * RH + j]);
      float hn = sigf(sm.gates[m][3 * RH + j]) * tanhf(cn);
      sm.cr1[m][j] = cn;
      sm.gin1[m][48 + j] = (_Float16)hn;
      outR[((long)(b0 + m) * T_LEN + t) * RH + j] = hn;
    }
    __syncthreads();

    // r_in = hr1 @ e1^T + e1_b -> catt[:,32:64]  (A read at kbase 48/80 of gin1)
    {
      v16h a0 = load_A(&sm.gin1[0][0], 128, 48);
      v16h a1 = load_A(&sm.gin1[0][0], 128, 80);
#pragma unroll
      for (int nt = 0; nt < 2; ++nt) {
        v8f acc = splat(be1[nt]);
        acc = wmma(a0, fe1[nt][0], acc);
        acc = wmma(a1, fe1[nt][1], acc);
        store_D_h(&sm.catt[0][0], 64, 32 + nt * 16, acc);
      }
    }
    __syncthreads();

    // in_t = [t_in | r_in] @ e4^T + e4_b -> itbuf (f32)
    {
      v16h a0 = load_A(&sm.catt[0][0], 64, 0);
      v16h a1 = load_A(&sm.catt[0][0], 64, 32);
#pragma unroll
      for (int nt = 0; nt < 2; ++nt) {
        v8f acc = splat(be4[nt]);
        acc = wmma(a0, fe4[nt][0], acc);
        acc = wmma(a1, fe4[nt][1], acc);
        store_D_f(&sm.itbuf[0][0], 32, nt * 16, acc);
      }
    }
    __syncthreads();

    // t-cell 0 (hidden 2): gates via VALU dot products
    for (int i = lane; i < MB * 8; i += 32) {
      int m = i >> 3, n = i & 7;
      float g = sm.tb0[n];
#pragma unroll
      for (int k = 0; k < 32; ++k) g += sm.itbuf[m][k] * sm.tw0[n][k];
      g += sm.ht0[m][0] * sm.twh0[n][0] + sm.ht0[m][1] * sm.twh0[n][1];
      sm.tg[m][n] = g;
    }
    __syncthreads();
    for (int i = lane; i < MB * 2; i += 32) {
      int m = i >> 1, j = i & 1;
      float cn = sigf(sm.tg[m][2 + j]) * sm.ct0[m][j]
               + sigf(sm.tg[m][j]) * tanhf(sm.tg[m][4 + j]);
      sm.ct0[m][j] = cn;
      sm.ht0[m][j] = sigf(sm.tg[m][6 + j]) * tanhf(cn);
    }
    __syncthreads();

    // t-cell 1
    for (int i = lane; i < MB * 8; i += 32) {
      int m = i >> 3, n = i & 7;
      sm.tg[m][n] = sm.tb1[n]
          + sm.ht0[m][0] * sm.twi1[n][0] + sm.ht0[m][1] * sm.twi1[n][1]
          + sm.ht1[m][0] * sm.twh1[n][0] + sm.ht1[m][1] * sm.twh1[n][1];
    }
    __syncthreads();
    for (int i = lane; i < MB * 2; i += 32) {
      int m = i >> 1, j = i & 1;
      float cn = sigf(sm.tg[m][2 + j]) * sm.ct1[m][j]
               + sigf(sm.tg[m][j]) * tanhf(sm.tg[m][4 + j]);
      sm.ct1[m][j] = cn;
      float hn = sigf(sm.tg[m][6 + j]) * tanhf(cn);
      sm.ht1[m][j] = hn;
      outT[((long)(b0 + m) * T_LEN + t) * TS + j] = hn;
    }
    __syncthreads();
  }
}

} // anonymous namespace

extern "C" void kernel_launch(void* const* d_in, const int* in_sizes, int n_in,
                              void* d_out, int out_size, void* d_ws, size_t ws_size,
                              hipStream_t stream) {
  (void)in_sizes; (void)n_in; (void)out_size; (void)d_ws; (void)ws_size;
  const float* p[26];
  for (int i = 0; i < 26; ++i) p[i] = (const float*)d_in[i];
  float* outR = (float*)d_out;
  float* outT = outR + (size_t)B_ALL * T_LEN * RH;

  // ~120 KB dynamic LDS per workgroup (gfx1250 WGP has 320 KB).
  hipFuncSetAttribute((const void*)adrnn_scan_kernel,
                      hipFuncAttributeMaxDynamicSharedMemorySize, (int)sizeof(SMem));

  adrnn_scan_kernel<<<B_ALL / MB, 32, sizeof(SMem), stream>>>(
      p[0], p[1], p[2], p[3], p[4], p[5], p[6], p[7], p[8], p[9],
      p[10], p[11], p[12], p[13], p[14], p[15], p[16], p[17],
      p[18], p[19], p[20], p[21], p[22], p[23], p[24], p[25],
      outR, outT);
}
